// PixelWiseNet_79328045957656
// MI455X (gfx1250) — matrix operand order repaired and checked by reading, not compiled, not run on hardware
//
#include <hip/hip_runtime.h>

// PixelWiseNet hinge-basis kernel for gfx1250 (MI455X).
// out[p] = bias + sum_{c,k} p_a[c,k] * relu(x[c,p]*rowsum[c] + bias_c[c] - p_t[c,k])
//
// Mapping: weights in WMMA A (all rows identical), per-pixel hinge activations
// in WMMA B (column = pixel). Two v_wmma_f32_16x16x32_f16 cover the 48 (padded
// to 64) hinge terms for 16 pixels; outputs land coalesced in acc[0] of lanes
// 0..15. Two tiles (32 pixels) are processed per iteration with independent
// accumulators to fill the WMMA->VALU hazard shadow and amortize loop control.

typedef _Float16 h2   __attribute__((ext_vector_type(2)));
typedef _Float16 v16h __attribute__((ext_vector_type(16)));
typedef float    v8f  __attribute__((ext_vector_type(8)));

#define HW_TOTAL (1024 * 1024)
#define N_TILES  (HW_TOTAL / 16)

__global__ __launch_bounds__(256) void hinge_wmma_kernel(
    const float* __restrict__ x,      // [3][H][W]
    const float* __restrict__ M,      // [3][3]
    const float* __restrict__ p_a,    // [3][16]
    const float* __restrict__ p_t,    // [3][16]
    const float* __restrict__ bias_c, // [3]
    const float* __restrict__ bias,   // [1]
    float* __restrict__ out)          // [H][W]
{
    const int lane = threadIdx.x & 31;
    const int half = lane >> 4;    // 0: lanes 0-15, 1: lanes 16-31
    const int n    = lane & 15;    // pixel-in-tile == WMMA column
    const int c01  = half;         // channel this lane computes for WMMA0's B
    const int kOff = half ? 8 : 0; // A-matrix K sub-block for this lane half

    // Per-channel affine: s_c = x_c * rowsum_c + bias_c
    float rs0 = M[0] + M[1] + M[2];
    float rs1 = M[3] + M[4] + M[5];
    float rs2 = M[6] + M[7] + M[8];
    const float rs01 = half ? rs1 : rs0;
    const float bc01 = half ? bias_c[1] : bias_c[0];
    const float bc2  = bias_c[2];
    const float biasOut = bias[0];

    // ---- Build A matrices (f16 weights, all 16 rows identical) ----
    // WMMA0 covers K=0..31  (c0 k0-15, c1 k0-15)
    // WMMA1 covers K=32..63 (c2 k0-15, then zeros for K>=48)
    // A k order used here (contiguous, matches B): lanes 0-15 hold K 0-15; lanes 16-31 K 16-31.
    union { v16h v; h2 p[8]; } A0, A1, B00, B10, B01, B11;
#pragma unroll
    for (int j = 0; j < 4; ++j) {
        A0.p[j]     = h2{(_Float16)p_a[c01 * 16 + 2 * j],
                         (_Float16)p_a[c01 * 16 + 2 * j + 1]};
        A0.p[4 + j] = h2{(_Float16)p_a[c01 * 16 + 8 + 2 * j],
                         (_Float16)p_a[c01 * 16 + 8 + 2 * j + 1]};
        A1.p[j]     = half ? h2{(_Float16)0.0f, (_Float16)0.0f}
                           : h2{(_Float16)p_a[2 * 16 + 2 * j],
                                (_Float16)p_a[2 * 16 + 2 * j + 1]};
        A1.p[4 + j] = half ? h2{(_Float16)0.0f, (_Float16)0.0f} // zero weights for padded K
                           : h2{(_Float16)p_a[2 * 16 + 8 + 2 * j],
                                (_Float16)p_a[2 * 16 + 8 + 2 * j + 1]};
    }

    // ---- Thresholds for this lane's B rows ----
    // B k order used here (contiguous, matches A): lanes 0-15 hold K 0-15, lanes 16-31 hold K 16-31; col = lane%16.
    // => WMMA0: this lane provides channel c01's 16 hinges for pixel n.
    //    WMMA1: lanes 0-15 provide channel 2's hinges; lanes 16-31 hit zero weights.
    h2 tA[8], tB[8];
#pragma unroll
    for (int j = 0; j < 8; ++j) {
        tA[j] = h2{(_Float16)p_t[c01 * 16 + 2 * j], (_Float16)p_t[c01 * 16 + 2 * j + 1]};
        tB[j] = h2{(_Float16)p_t[2 * 16 + 2 * j],   (_Float16)p_t[2 * 16 + 2 * j + 1]};
    }

    const float* __restrict__ x01 = x + (size_t)c01 * HW_TOTAL;
    const float* __restrict__ x2p = x + (size_t)2 * HW_TOTAL;

    const int wavesTotal = (gridDim.x * blockDim.x) >> 5;
    const int wave   = (blockIdx.x * blockDim.x + threadIdx.x) >> 5;
    const int stride = wavesTotal * 2;            // tiles advanced per iteration
    const h2 zero = h2{(_Float16)0.0f, (_Float16)0.0f};

    // N_TILES (65536) divides exactly by stride for the launch below: no tail,
    // both tiles of each pair always valid, EXEC all-1s at every WMMA.
    for (int t = wave * 2; t < N_TILES; t += stride) {
        const int idx0 = t * 16 + n;
        const int idx1 = idx0 + 16;

        // 128B contiguous per plane per wave per iteration.
        const float xa0 = x01[idx0];
        const float xa1 = x01[idx1];
        const float xb0 = x2p[idx0];
        const float xb1 = x2p[idx1];

        // Prefetch next iteration's pair (wave-uniform guard; streaming data).
        const int pf = idx0 + stride * 16;
        if (t + stride < N_TILES) {
            __builtin_prefetch(x01 + pf, 0, 1);
            __builtin_prefetch(x01 + pf + 16, 0, 1);
            __builtin_prefetch(x2p + pf, 0, 1);
            __builtin_prefetch(x2p + pf + 16, 0, 1);
        }

        const float s010 = fmaf(xa0, rs01, bc01);
        const float s011 = fmaf(xa1, rs01, bc01);
        const float s20  = fmaf(xb0, rs2, bc2);
        const float s21  = fmaf(xb1, rs2, bc2);
        const h2 sh00 = h2{(_Float16)s010, (_Float16)s010};
        const h2 sh01 = h2{(_Float16)s011, (_Float16)s011};
        const h2 sh20 = h2{(_Float16)s20, (_Float16)s20};
        const h2 sh21 = h2{(_Float16)s21, (_Float16)s21};

        // Hinge construction: packed f16 sub + max (v_pk_add_f16 / v_pk_max_num_f16).
#pragma unroll
        for (int j = 0; j < 8; ++j) {
            B00.p[j] = __builtin_elementwise_max(sh00 - tA[j], zero);
            B01.p[j] = __builtin_elementwise_max(sh01 - tA[j], zero);
            B10.p[j] = __builtin_elementwise_max(sh20 - tB[j], zero);
            B11.p[j] = __builtin_elementwise_max(sh21 - tB[j], zero);
        }

        // Interleave the two tiles' WMMAs: no WMMA immediately feeds a
        // dependent WMMA/VALU, so hazard NOPs are filled with real work.
        v8f acc0 = {};
        v8f acc1 = {};
        acc0 = __builtin_amdgcn_wmma_f32_16x16x32_f16(
            false, A0.v, false, B00.v, (short)0, acc0, false, false);
        acc1 = __builtin_amdgcn_wmma_f32_16x16x32_f16(
            false, A0.v, false, B01.v, (short)0, acc1, false, false);
        acc0 = __builtin_amdgcn_wmma_f32_16x16x32_f16(
            false, A1.v, false, B10.v, (short)0, acc0, false, false);
        acc1 = __builtin_amdgcn_wmma_f32_16x16x32_f16(
            false, A1.v, false, B11.v, (short)0, acc1, false, false);

        // D[0][n] sits in acc element 0 of lane n (lanes 0-15): two coalesced
        // 64B stores (128B contiguous).
        if (lane < 16) {
            out[idx0] = acc0[0] + biasOut;
            out[idx1] = acc1[0] + biasOut;
        }
    }
}

extern "C" void kernel_launch(void* const* d_in, const int* in_sizes, int n_in,
                              void* d_out, int out_size, void* d_ws, size_t ws_size,
                              hipStream_t stream) {
    (void)in_sizes; (void)n_in; (void)d_ws; (void)ws_size; (void)out_size;
    const float* x      = (const float*)d_in[0];
    const float* M      = (const float*)d_in[1];
    const float* p_a    = (const float*)d_in[2];
    const float* p_t    = (const float*)d_in[3];
    const float* bias_c = (const float*)d_in[4];
    const float* bias   = (const float*)d_in[5];
    float* out = (float*)d_out;

    // 1024 blocks x 256 threads = 8192 waves; each wave handles 4 iterations
    // of 2 tiles (32 pixels) => 65536 tiles total, exact coverage.
    hinge_wmma_kernel<<<1024, 256, 0, stream>>>(x, M, p_a, p_t, bias_c, bias, out);
}
